// DeformableDecoderLayer_79869211836551
// MI455X (gfx1250) — matrix operand order repaired and checked by reading, not compiled, not run on hardware
//
#include <hip/hip_runtime.h>

typedef __attribute__((ext_vector_type(16))) _Float16 v16h;
typedef __attribute__((ext_vector_type(8)))  float    v8f;

#define WMMA_F16(a, b, c) \
    __builtin_amdgcn_wmma_f32_16x16x32_f16(false, (a), false, (b), (short)0, (c), false, false)

// ---------------------------------------------------------------------------
// Generic GEMM: C[M,N] = act(A[M,K] @ W[N,K]^T + bias[N]) ; f32 in/out,
// f16 WMMA compute. One wave computes one 32x32 C tile (4 WMMAs / K-step,
// 2x operand reuse). block = 128 (4 waves); grid = (N/32, ceil(M/128)).
// ---------------------------------------------------------------------------
template <bool RELU>
__global__ __launch_bounds__(128) void gemm_wmma(
    const float* __restrict__ A, const float* __restrict__ W,
    const float* __restrict__ bias, float* __restrict__ C,
    int M, int N, int K)
{
    const int lane = threadIdx.x & 31;
    const int wave = threadIdx.x >> 5;
    const int n0 = blockIdx.x * 32;
    const int m0 = (blockIdx.y * 4 + wave) * 32;
    if (m0 >= M) return;                       // wave-uniform exit

    const int lr  = lane & 15;
    const int hs  = lane >> 4;
    const int g   = hs * 8;                    // A-frag K-group base
    const int h16 = hs * 16;                   // W-row prefetch K offset

    int ra0 = m0 + lr;                         // clamp rows for ragged M (M=144)
    int ra1 = m0 + 16 + lr;
    if (ra0 >= M) ra0 = M - 1;
    if (ra1 >= M) ra1 = M - 1;
    const float* arow0 = A + (size_t)ra0 * K;
    const float* arow1 = A + (size_t)ra1 * K;
    const float* wrow0 = W + (size_t)(n0 + lr) * K;
    const float* wrow1 = W + (size_t)(n0 + 16 + lr) * K;

    v8f acc00 = {}, acc01 = {}, acc10 = {}, acc11 = {};
    for (int k0 = 0; k0 < K; k0 += 32) {
        // speculative prefetch of next K-step operands (global_prefetch_b8)
        if (k0 + 32 < K) {
            __builtin_prefetch(arow0 + k0 + 32 + g, 0, 0);
            __builtin_prefetch(arow1 + k0 + 32 + g, 0, 0);
            __builtin_prefetch(wrow0 + k0 + 32 + h16, 0, 0);
            __builtin_prefetch(wrow1 + k0 + 32 + h16, 0, 0);
        }
        v16h a0, a1, b0, b1;
#pragma unroll
        for (int j = 0; j < 8; ++j) {
            a0[j]     = (_Float16)arow0[k0 + g + j];
            a0[j + 8] = (_Float16)arow0[k0 + 16 + g + j];
            a1[j]     = (_Float16)arow1[k0 + g + j];
            a1[j + 8] = (_Float16)arow1[k0 + 16 + g + j];
        }
#pragma unroll
        for (int j = 0; j < 16; ++j) {
            b0[j] = (_Float16)wrow0[k0 + (j < 8 ? g + j : 8 + g + j)];
            b1[j] = (_Float16)wrow1[k0 + (j < 8 ? g + j : 8 + g + j)];
        }
        acc00 = WMMA_F16(a0, b0, acc00);
        acc01 = WMMA_F16(a0, b1, acc01);
        acc10 = WMMA_F16(a1, b0, acc10);
        acc11 = WMMA_F16(a1, b1, acc11);
    }

    const int   c0  = n0 + lr, c1 = n0 + 16 + lr;
    const float bv0 = bias ? bias[c0] : 0.0f;
    const float bv1 = bias ? bias[c1] : 0.0f;
#pragma unroll
    for (int r = 0; r < 8; ++r) {
        int row = m0 + r + hs * 8;
        if (row < M) {
            float v = acc00[r] + bv0; if (RELU) v = fmaxf(v, 0.0f);
            C[(size_t)row * N + c0] = v;
            v = acc01[r] + bv1; if (RELU) v = fmaxf(v, 0.0f);
            C[(size_t)row * N + c1] = v;
        }
        int row2 = row + 16;
        if (row2 < M) {
            float v = acc10[r] + bv0; if (RELU) v = fmaxf(v, 0.0f);
            C[(size_t)row2 * N + c0] = v;
            v = acc11[r] + bv1; if (RELU) v = fmaxf(v, 0.0f);
            C[(size_t)row2 * N + c1] = v;
        }
    }
}

// ---------------------------------------------------------------------------
// Flash attention, 8 heads, DH=32. One wave handles one (b, h, 16-query tile).
// Softmax is done in the A-fragment domain: raw f32 scores are bounced through
// LDS (a bounce needed anyway to build the P fragment); each lane then owns a
// half-row, so row max/sum need exactly one shfl_xor(16) each. Per-row rescale
// factors are broadcast back to the C-layout accumulator via a 16-float LDS
// array (read as 2 x b128). No serialized bpermute butterflies.
// block = 128 (4 waves, each its own LDS tile); grid = B*8*(Sq/16)/4 blocks.
// ---------------------------------------------------------------------------
__global__ __launch_bounds__(128) void flash_attn(
    const float* __restrict__ Q, int ldq, int qcol,
    const float* __restrict__ Km, int ldk, int kcol,
    const float* __restrict__ Vm, int ldv, int vcol,
    float* __restrict__ O, int ldo,
    int Sq, int Skv, float scale)
{
    __shared__ float ldsS[4][16][33];   // raw scores, padded rows (bank spread)
    __shared__ float rsbuf[4][16];      // per-row rescale / inv-sum broadcast

    const int lane = threadIdx.x & 31;
    const int wave = threadIdx.x >> 5;
    const int qtiles = Sq / 16;
    const int gw = blockIdx.x * 4 + wave;
    const int qtile = gw % qtiles;
    const int h = (gw / qtiles) & 7;
    const int b = gw / (qtiles * 8);

    const int lr  = lane & 15;
    const int hs  = lane >> 4;
    const int g   = hs * 8;
    const int h16 = hs * 16;

    // Q fragment (A layout), pre-scaled by 1/sqrt(DH)
    v16h qa;
    {
        const float* qrow = Q + ((size_t)b * Sq + qtile * 16 + lr) * ldq + qcol + h * 32;
#pragma unroll
        for (int j = 0; j < 8; ++j) {
            qa[j]     = (_Float16)(qrow[g + j] * scale);
            qa[j + 8] = (_Float16)(qrow[16 + g + j] * scale);
        }
    }

    float mrun = -1e30f, lrun = 0.0f;   // running stats for row lr (A-domain)
    v8f acc0 = {}, acc1 = {};

    const int nkt = (Skv + 31) / 32;
    for (int t = 0; t < nkt; ++t) {
        // ---- K fragments (B layout): S0 = keys [t*32, +16), S1 = [+16, +32)
        v16h kb0, kb1;
        {
            int k0i = t * 32 + lr,      k1i = k0i + 16;
            int c0  = k0i < Skv ? k0i : Skv - 1;
            int c1  = k1i < Skv ? k1i : Skv - 1;
            const float* kr0 = Km + ((size_t)b * Skv + c0) * ldk + kcol + h * 32;
            const float* kr1 = Km + ((size_t)b * Skv + c1) * ldk + kcol + h * 32;
#pragma unroll
            for (int j = 0; j < 16; ++j) {
                kb0[j] = (_Float16)kr0[j < 8 ? g + j : 8 + g + j];
                kb1[j] = (_Float16)kr1[j < 8 ? g + j : 8 + g + j];
            }
        }
        // prefetch next KV tile's K rows while this tile computes
        if (t + 1 < nkt) {
            int nk = t * 32 + 32 + lr;
            if (nk >= Skv) nk = Skv - 1;
            __builtin_prefetch(Km + ((size_t)b * Skv + nk) * ldk + kcol + h * 32 + h16, 0, 0);
        }
        v8f z0 = {}, z1 = {};
        v8f s0 = WMMA_F16(qa, kb0, z0);
        v8f s1 = WMMA_F16(qa, kb1, z1);

        const bool ok0 = (t * 32 + lr) < Skv;
        const bool ok1 = (t * 32 + 16 + lr) < Skv;

        // ---- bounce masked raw scores (f32) through LDS: C-layout -> A-layout
        __syncthreads();   // previous iteration's LDS reads complete
#pragma unroll
        for (int r = 0; r < 8; ++r) {
            int m = r + hs * 8;
            ldsS[wave][m][lr]      = ok0 ? s0[r] : -1e30f;
            ldsS[wave][m][16 + lr] = ok1 ? s1[r] : -1e30f;
        }
        __syncthreads();

        // ---- A-domain softmax: this lane owns half of row lr
        float x[16];
#pragma unroll
        for (int j = 0; j < 8; ++j) {
            x[j]     = ldsS[wave][lr][g + j];
            x[j + 8] = ldsS[wave][lr][16 + g + j];
        }
        float pm = x[0];
#pragma unroll
        for (int j = 1; j < 16; ++j) pm = fmaxf(pm, x[j]);
        pm = fmaxf(pm, __shfl_xor(pm, 16));            // full row max
        float mnew = fmaxf(mrun, pm);
        float resc = __expf(mrun - mnew);
        mrun = mnew;

        v16h pa;
        float ps = 0.0f;
#pragma unroll
        for (int j = 0; j < 16; ++j) {
            float p = __expf(x[j] - mnew);
            ps += p;
            pa[j] = (_Float16)p;
        }
        ps += __shfl_xor(ps, 16);                      // full row sum
        lrun = lrun * resc + ps;

        // ---- broadcast per-row rescale to C-layout accumulator lanes
        rsbuf[wave][lr] = resc;        // both half-lanes write identical value
        __syncthreads();
#pragma unroll
        for (int r = 0; r < 8; ++r) {
            float rc = rsbuf[wave][r + hs * 8];
            acc0[r] *= rc;
            acc1[r] *= rc;
        }

        // ---- V fragments (B layout): columns = head dims [0,16) and [16,32)
        v16h vb0, vb1;
#pragma unroll
        for (int j = 0; j < 16; ++j) {
            int kk = t * 32 + (j < 8 ? g + j : 8 + g + j);
            int ck = kk < Skv ? kk : Skv - 1;
            const float* vr = Vm + ((size_t)b * Skv + ck) * ldv + vcol + h * 32;
            vb0[j] = (_Float16)vr[lr];
            vb1[j] = (_Float16)vr[16 + lr];
        }
        acc0 = WMMA_F16(pa, vb0, acc0);
        acc1 = WMMA_F16(pa, vb1, acc1);
    }

    // ---- broadcast per-row 1/sum and store normalized output
    __syncthreads();
    rsbuf[wave][lr] = 1.0f / lrun;
    __syncthreads();
#pragma unroll
    for (int r = 0; r < 8; ++r) {
        int   m   = r + hs * 8;
        int   row = qtile * 16 + m;
        float inv = rsbuf[wave][m];
        float* orow = O + ((size_t)b * Sq + row) * ldo + h * 32;
        orow[lr]      = acc0[r] * inv;
        orow[16 + lr] = acc1[r] * inv;
    }
}

// ---------------------------------------------------------------------------
// out = LayerNorm(x + res) over D=256; one wave per row.
// ---------------------------------------------------------------------------
__global__ __launch_bounds__(256) void residual_ln(
    const float* __restrict__ x, const float* __restrict__ res,
    const float* __restrict__ gamma, const float* __restrict__ beta,
    float* __restrict__ out, int rows)
{
    const int row  = blockIdx.x * 8 + (threadIdx.x >> 5);
    const int lane = threadIdx.x & 31;
    if (row >= rows) return;
    const float* xr = x   + (size_t)row * 256;
    const float* rr = res + (size_t)row * 256;

    float v[8], s = 0.0f, s2 = 0.0f;
#pragma unroll
    for (int j = 0; j < 8; ++j) {
        v[j] = xr[lane + 32 * j] + rr[lane + 32 * j];
        s  += v[j];
        s2 += v[j] * v[j];
    }
#pragma unroll
    for (int m = 1; m < 32; m <<= 1) {
        s  += __shfl_xor(s, m);
        s2 += __shfl_xor(s2, m);
    }
    float mean = s * (1.0f / 256.0f);
    float var  = s2 * (1.0f / 256.0f) - mean * mean;
    float inv  = rsqrtf(var + 1e-5f);
#pragma unroll
    for (int j = 0; j < 8; ++j) {
        int c = lane + 32 * j;
        out[(size_t)row * 256 + c] = (v[j] - mean) * inv * gamma[c] + beta[c];
    }
}

// ---------------------------------------------------------------------------
// Deformable multi-scale sampling. One wave per (b, q, h); lane = channel d.
// off: (B*Nq,192), aw: (B*Nq,96) logits, values v_l: (B*H*W,256).
// ---------------------------------------------------------------------------
__global__ __launch_bounds__(256) void deform_sample(
    const float* __restrict__ refp, const float* __restrict__ off,
    const float* __restrict__ awl,
    const float* __restrict__ v0, const float* __restrict__ v1,
    const float* __restrict__ v2,
    float* __restrict__ outv, int B, int Nq)
{
    const int gw   = blockIdx.x * 8 + (threadIdx.x >> 5);
    const int lane = threadIdx.x & 31;
    const int h  = gw & 7;
    const int bq = gw >> 3;
    if (bq >= B * Nq) return;
    const int b = bq / Nq;

    const float rx = refp[(size_t)bq * 2 + 0];
    const float ry = refp[(size_t)bq * 2 + 1];

    // softmax over NL*NP = 12 attention logits for this head
    const float* ar = awl + (size_t)bq * 96 + h * 12;
    float mx = -1e30f;
#pragma unroll
    for (int i = 0; i < 12; ++i) mx = fmaxf(mx, ar[i]);
    float w[12], se = 0.0f;
#pragma unroll
    for (int i = 0; i < 12; ++i) { w[i] = __expf(ar[i] - mx); se += w[i]; }
    const float invse = 1.0f / se;

    const float* orow = off + (size_t)bq * 192 + h * 24;
    const float* vs[3] = { v0, v1, v2 };
    const int    hw[3] = { 80, 40, 20 };

    float acc = 0.0f;
#pragma unroll
    for (int l = 0; l < 3; ++l) {
        const int    Wd = hw[l], Hd = hw[l];
        const float* V  = vs[l];
#pragma unroll
        for (int p = 0; p < 4; ++p) {
            float ox = orow[(l * 4 + p) * 2 + 0];
            float oy = orow[(l * 4 + p) * 2 + 1];
            float x = (rx + ox / (float)Wd) * (float)Wd - 0.5f;
            float y = (ry + oy / (float)Hd) * (float)Hd - 0.5f;
            float x0f = floorf(x), y0f = floorf(y);
            float wx = x - x0f, wy = y - y0f;
            int x0 = (int)x0f, y0 = (int)y0f;
            float samp = 0.0f;
#pragma unroll
            for (int cy = 0; cy < 2; ++cy)
#pragma unroll
                for (int cx = 0; cx < 2; ++cx) {
                    int ix = x0 + cx, iy = y0 + cy;
                    if (ix >= 0 && ix < Wd && iy >= 0 && iy < Hd) {
                        float cw = (cx ? wx : 1.0f - wx) * (cy ? wy : 1.0f - wy);
                        size_t idx = (size_t)b * (Hd * Wd) + (size_t)iy * Wd + ix;
                        samp += cw * V[idx * 256 + h * 32 + lane];
                    }
                }
            acc += (w[l * 4 + p] * invse) * samp;
        }
    }
    outv[(size_t)bq * 256 + h * 32 + lane] = acc;
}

// ---------------------------------------------------------------------------
extern "C" void kernel_launch(void* const* d_in, const int* in_sizes, int n_in,
                              void* d_out, int out_size, void* d_ws, size_t ws_size,
                              hipStream_t stream)
{
    (void)in_sizes; (void)n_in; (void)out_size; (void)ws_size;

    const float* query    = (const float*)d_in[0];
    const float* templ    = (const float*)d_in[1];
    const float* feat0    = (const float*)d_in[2];
    const float* feat1    = (const float*)d_in[3];
    const float* feat2    = (const float*)d_in[4];
    const float* refpts   = (const float*)d_in[5];
    const float* sa_w_in  = (const float*)d_in[6];
    const float* sa_b_in  = (const float*)d_in[7];
    const float* sa_w_out = (const float*)d_in[8];
    const float* sa_b_out = (const float*)d_in[9];
    const float* ta_w_in  = (const float*)d_in[10];
    const float* ta_b_in  = (const float*)d_in[11];
    const float* ta_w_out = (const float*)d_in[12];
    const float* ta_b_out = (const float*)d_in[13];
    const float* off_w    = (const float*)d_in[14];
    const float* off_b    = (const float*)d_in[15];
    const float* aw_w     = (const float*)d_in[16];
    const float* aw_b     = (const float*)d_in[17];
    const float* val_w    = (const float*)d_in[18];
    const float* val_b    = (const float*)d_in[19];
    const float* dout_w   = (const float*)d_in[20];
    const float* dout_b   = (const float*)d_in[21];
    const float* ffn_w1   = (const float*)d_in[22];
    const float* ffn_b1   = (const float*)d_in[23];
    const float* ffn_w2   = (const float*)d_in[24];
    const float* ffn_b2   = (const float*)d_in[25];
    const float* ln1_g = (const float*)d_in[26]; const float* ln1_b = (const float*)d_in[27];
    const float* ln2_g = (const float*)d_in[28]; const float* ln2_b = (const float*)d_in[29];
    const float* ln3_g = (const float*)d_in[30]; const float* ln3_b = (const float*)d_in[31];
    const float* ln4_g = (const float*)d_in[32]; const float* ln4_b = (const float*)d_in[33];
    float* out = (float*)d_out;

    const int B = 16, Nq = 512, ROWS = B * Nq;   // 8192
    const float scale = 0.1767766952966369f;      // 1/sqrt(32)

    // ---- workspace carve (bytes, 256-aligned), with reuse ----
    size_t cur = 0;
    auto carve = [&](size_t nfloats) -> float* {
        float* p = (float*)((char*)d_ws + cur);
        cur += nfloats * sizeof(float);
        cur = (cur + 255) & ~(size_t)255;
        return p;
    };
    float* W_big = carve((size_t)ROWS * 1024);  // qkv (8192x768) then ffn1 (8192x1024)
    float* W_a   = carve((size_t)ROWS * 256);   // attn-out / attn2-out / dval
    float* W_b   = carve((size_t)ROWS * 256);   // proj / proj2 / dproj / ffn2
    float* W_q1  = carve((size_t)ROWS * 256);   // q1 then q3
    float* W_q2  = carve((size_t)ROWS * 256);   // q_ta then q2
    float* W_kv  = carve((size_t)144 * 512);    // template KV
    float* W_v0  = carve((size_t)B * 6400 * 256);
    float* W_v1  = carve((size_t)B * 1600 * 256);
    float* W_v2  = carve((size_t)B * 400 * 256);
    float* W_off = carve((size_t)ROWS * 192);
    float* W_aw  = carve((size_t)ROWS * 96);

    const dim3 blk128(128), blk256(256);
    const int  MB = ROWS / 128;                 // 64 m-blocks for 8192 rows
    const dim3 gN256(8, MB);

    // 1. self-attn QKV projection (8192 x 768)
    gemm_wmma<false><<<dim3(24, MB), blk128, 0, stream>>>(
        query, sa_w_in, sa_b_in, W_big, ROWS, 768, 256);
    // 2. self-attention (Skv = 512)
    flash_attn<<<dim3(B * 8 * (Nq / 16) / 4), blk128, 0, stream>>>(
        W_big, 768, 0, W_big, 768, 256, W_big, 768, 512,
        W_a, 256, Nq, Nq, scale);
    // 3. self-attn output projection
    gemm_wmma<false><<<gN256, blk128, 0, stream>>>(W_a, sa_w_out, sa_b_out, W_b, ROWS, 256, 256);
    // 4. LN1
    residual_ln<<<dim3(ROWS / 8), blk256, 0, stream>>>(W_b, query, ln1_g, ln1_b, W_q1, ROWS);

    // 5. template cross-attn: Q projection (rows 0..255 of packed in-proj)
    gemm_wmma<false><<<gN256, blk128, 0, stream>>>(W_q1, ta_w_in, ta_b_in, W_q2, ROWS, 256, 256);
    // 6. template KV projection (144 x 512, weight rows 256..767)
    gemm_wmma<false><<<dim3(16, 2), blk128, 0, stream>>>(
        templ, ta_w_in + (size_t)256 * 256, ta_b_in + 256, W_kv, 144, 512, 256);
    // 7. cross-attention (Skv = 9)
    flash_attn<<<dim3(B * 8 * (Nq / 16) / 4), blk128, 0, stream>>>(
        W_q2, 256, 0, W_kv, 512, 0, W_kv, 512, 256,
        W_a, 256, Nq, 9, scale);
    // 8. cross-attn output projection
    gemm_wmma<false><<<gN256, blk128, 0, stream>>>(W_a, ta_w_out, ta_b_out, W_b, ROWS, 256, 256);
    // 9. LN2
    residual_ln<<<dim3(ROWS / 8), blk256, 0, stream>>>(W_b, W_q1, ln2_g, ln2_b, W_q2, ROWS);

    // 10-11. value projections of the three feature levels
    gemm_wmma<false><<<dim3(8, 800), blk128, 0, stream>>>(feat0, val_w, val_b, W_v0, B * 6400, 256, 256);
    gemm_wmma<false><<<dim3(8, 200), blk128, 0, stream>>>(feat1, val_w, val_b, W_v1, B * 1600, 256, 256);
    gemm_wmma<false><<<dim3(8, 50),  blk128, 0, stream>>>(feat2, val_w, val_b, W_v2, B * 400,  256, 256);
    // 12. sampling-offset projection (N=192) and 13. attention-weight logits (N=96)
    gemm_wmma<false><<<dim3(6, MB), blk128, 0, stream>>>(W_q2, off_w, off_b, W_off, ROWS, 192, 256);
    gemm_wmma<false><<<dim3(3, MB), blk128, 0, stream>>>(W_q2, aw_w, aw_b, W_aw, ROWS, 96, 256);
    // 14. deformable bilinear sampling + weighted sum -> (8192, 256)
    deform_sample<<<dim3(ROWS * 8 / 8), blk256, 0, stream>>>(
        refpts, W_off, W_aw, W_v0, W_v1, W_v2, W_a, B, Nq);
    // 15. deformable output projection
    gemm_wmma<false><<<gN256, blk128, 0, stream>>>(W_a, dout_w, dout_b, W_b, ROWS, 256, 256);
    // 16. LN3
    residual_ln<<<dim3(ROWS / 8), blk256, 0, stream>>>(W_b, W_q2, ln3_g, ln3_b, W_q1, ROWS);

    // 17. FFN up (ReLU) and 18. FFN down
    gemm_wmma<true><<<dim3(32, MB), blk128, 0, stream>>>(W_q1, ffn_w1, ffn_b1, W_big, ROWS, 1024, 256);
    gemm_wmma<false><<<gN256, blk128, 0, stream>>>(W_big, ffn_w2, ffn_b2, W_b, ROWS, 256, 1024);
    // 19. LN4 -> final output
    residual_ln<<<dim3(ROWS / 8), blk256, 0, stream>>>(W_b, W_q1, ln4_g, ln4_b, out, ROWS);
}
